// GlobalReadoutBlock_80315888435956
// MI455X (gfx1250) — matrix operand order repaired and checked by reading, not compiled, not run on hardware
//
#include <hip/hip_runtime.h>
#include <hip/hip_bf16.h>

// ---------------------------------------------------------------------------
// Types / WMMA helpers (CDNA5 / gfx1250, wave32)
// ---------------------------------------------------------------------------
typedef __bf16 bf16_t;
typedef __bf16 v16bf __attribute__((ext_vector_type(16)));
typedef float  v8f   __attribute__((ext_vector_type(8)));

union FragB16 { v16bf v; unsigned u[8]; };

__device__ __forceinline__ v8f wmma_bf16(const FragB16 &a, const FragB16 &b, v8f c) {
  // D = A(16x32) * B(32x16) + C, f32 accumulate
  return __builtin_amdgcn_wmma_f32_16x16x32_bf16(false, a.v, false, b.v,
                                                 (short)0, c, false, false);
}

// ds_swizzle_b32 XOR-lane reductions (imm: xor<<10 | and_mask 0x1f)
#define SWZF(v, imm) __int_as_float(__builtin_amdgcn_ds_swizzle(__float_as_int(v), (imm)))
__device__ __forceinline__ float red_max16(float v) {
  v = fmaxf(v, SWZF(v, 0x041f)); v = fmaxf(v, SWZF(v, 0x081f));
  v = fmaxf(v, SWZF(v, 0x101f)); v = fmaxf(v, SWZF(v, 0x201f));
  return v;
}
__device__ __forceinline__ float red_add16(float v) {
  v += SWZF(v, 0x041f); v += SWZF(v, 0x081f);
  v += SWZF(v, 0x101f); v += SWZF(v, 0x201f);
  return v;
}
__device__ __forceinline__ float red_add32(float v) {
  v = red_add16(v); v += SWZF(v, 0x401f); return v;
}

// CDNA5 async copy: global -> LDS, 16B per lane, tracked by ASYNCcnt
__device__ __forceinline__ void async_copy_b128(const bf16_t* gp, const bf16_t* lp) {
  unsigned lds = (unsigned)(unsigned long long)lp;          // low 32b = LDS offset
  asm volatile("global_load_async_to_lds_b128 %0, %1, off"
               :: "v"(lds), "v"((unsigned long long)gp) : "memory");
}
__device__ __forceinline__ void wait_async0() {
  asm volatile("s_wait_asynccnt 0x0" ::: "memory");
}
__device__ __forceinline__ void wait_async3() {   // allow 3 newest in flight
  asm volatile("s_wait_asynccnt 0x3" ::: "memory");
}

// ---------------------------------------------------------------------------
// Elementwise f32 -> bf16 convert
// ---------------------------------------------------------------------------
__global__ void cvt_f32_bf16(const float* __restrict__ in, bf16_t* __restrict__ out,
                             long long n) {
  long long i = (long long)blockIdx.x * blockDim.x + threadIdx.x;
  long long stride = (long long)gridDim.x * blockDim.x;
  for (; i < n; i += stride) out[i] = (bf16_t)in[i];
}

// f32 (L,K,N) -> bf16 transposed (L,N,K): Wt[l][n][k] = W[l][k][n]
__global__ void cvtT_f32_bf16(const float* __restrict__ in, bf16_t* __restrict__ out,
                              int K, int N, long long total) {
  long long i = (long long)blockIdx.x * blockDim.x + threadIdx.x;
  long long stride = (long long)gridDim.x * blockDim.x;
  const long long KN = (long long)K * N;
  for (; i < total; i += stride) {
    long long l = i / KN, r = i - l * KN;
    int n = (int)(r / K), k = (int)(r - (long long)n * K);
    out[i] = (bf16_t)in[l * KN + (long long)k * N + n];
  }
}

// ---------------------------------------------------------------------------
// Row LayerNorm: fp32 in -> bf16 out. One row per wave (wave32).
// ---------------------------------------------------------------------------
__global__ __launch_bounds__(256) void ln_rows_kernel(
    const float* __restrict__ x, const float* __restrict__ g,
    const float* __restrict__ b, bf16_t* __restrict__ out, int D, int rows) {
  const int wave = threadIdx.x >> 5, lane = threadIdx.x & 31;
  const int row = blockIdx.x * 8 + wave;
  if (row >= rows) return;
  const float* xr = x + (size_t)row * D;
  float s = 0.f, s2 = 0.f;
  for (int i = lane; i < D; i += 32) { float v = xr[i]; s += v; s2 += v * v; }
  s = red_add32(s); s2 = red_add32(s2);
  const float mean = s / D;
  const float var  = s2 / D - mean * mean;
  const float rstd = rsqrtf(var + 1e-5f);
  for (int i = lane; i < D; i += 32) {
    float v = (xr[i] - mean) * rstd * g[i] + b[i];
    out[(size_t)row * D + i] = (bf16_t)v;
  }
}

// ---------------------------------------------------------------------------
// WMMA GEMM: out = act(A(bf16,MxK) @ Wt(bf16,NxK)^T + bias) [+ resid(f32)]
// Block tile 64(M) x 128(N), 8 waves (2x4), wave tile 32x32, K step 32.
// Double-buffered async staging: each wave issues 3 async b128 copies per
// tile; s_wait_asynccnt 3 overlaps tile k+1 copies with tile k WMMAs.
// ACT: 0 none, 1 SiLU, 2 exact GELU.
// ---------------------------------------------------------------------------
#define BM 64
#define BN 128
#define BK 32
#define TSTR 40   // LDS tile stride in bf16: 80B rows -> 16B aligned, conflict-free

template<int ACT, bool OUT_BF16, bool RESID>
__global__ __launch_bounds__(256) void gemm_bf16_kernel(
    const bf16_t* __restrict__ A, const bf16_t* __restrict__ Wt,
    const float* __restrict__ bias, const float* __restrict__ resid,
    float* __restrict__ outF, bf16_t* __restrict__ outB,
    int M, int N, int K) {
  __shared__ bf16_t As[2][BM * TSTR];
  __shared__ bf16_t Bs[2][BN * TSTR];

  const int t = threadIdx.x;
  const int wave = t >> 5, lane = t & 31;
  const int g = lane >> 4, l16 = lane & 15;
  const int wm = wave & 1, wn = wave >> 1;           // 2 x 4 wave grid
  const int m0 = blockIdx.y * BM;
  const int n0 = blockIdx.x * BN;

  const int rowa = t >> 2, sega = t & 3;             // A: 1 b128 per thread
  const int rowb0 = t >> 2, rowb1 = (256 + t) >> 2;  // B: 2 b128 per thread
  const int segb = t & 3;

  v8f acc[2][2];
#pragma unroll
  for (int i = 0; i < 2; ++i)
#pragma unroll
    for (int j = 0; j < 2; ++j) acc[i][j] = v8f{0, 0, 0, 0, 0, 0, 0, 0};

  // prologue: stage tile 0 into buffer 0
  async_copy_b128(A + (size_t)(m0 + rowa) * K + sega * 8, &As[0][rowa * TSTR + sega * 8]);
  async_copy_b128(Wt + (size_t)(n0 + rowb0) * K + segb * 8, &Bs[0][rowb0 * TSTR + segb * 8]);
  async_copy_b128(Wt + (size_t)(n0 + rowb1) * K + segb * 8, &Bs[0][rowb1 * TSTR + segb * 8]);

  const int NT = K / BK;
  for (int kt = 0; kt < NT; ++kt) {
    const int cur = kt & 1;
    if (kt + 1 < NT) {
      const int nxt = cur ^ 1;
      const int k1 = (kt + 1) * BK;
      async_copy_b128(A + (size_t)(m0 + rowa) * K + k1 + sega * 8,
                      &As[nxt][rowa * TSTR + sega * 8]);
      async_copy_b128(Wt + (size_t)(n0 + rowb0) * K + k1 + segb * 8,
                      &Bs[nxt][rowb0 * TSTR + segb * 8]);
      async_copy_b128(Wt + (size_t)(n0 + rowb1) * K + k1 + segb * 8,
                      &Bs[nxt][rowb1 * TSTR + segb * 8]);
      wait_async3();          // tile kt landed; tile kt+1 still in flight
    } else {
      wait_async0();
    }
    __syncthreads();

    // --- fragments (ISA 7.12.2 16-bit layouts) ---
    FragB16 a[2], b[2];
#pragma unroll
    for (int mt = 0; mt < 2; ++mt) {
      const int row = wm * 32 + mt * 16 + l16;
#pragma unroll
      for (int j = 0; j < 8; ++j) {
        const int kb = 2 * (j & 3) + 16 * (j >> 2) + 8 * g;     // A: 16x32 map
        a[mt].u[j] = *(const unsigned*)&As[cur][row * TSTR + kb];
      }
    }
#pragma unroll
    for (int nt = 0; nt < 2; ++nt) {
      const int col = wn * 32 + nt * 16 + l16;
#pragma unroll
      for (int j = 0; j < 8; ++j) {
        const int kb = 2 * (j & 3) + 16 * (j >> 2) + 8 * g;     // B: 32x16 map
        b[nt].u[j] = *(const unsigned*)&Bs[cur][col * TSTR + kb];
      }
    }
#pragma unroll
    for (int mt = 0; mt < 2; ++mt)
#pragma unroll
      for (int nt = 0; nt < 2; ++nt)
        acc[mt][nt] = wmma_bf16(a[mt], b[nt], acc[mt][nt]);
    __syncthreads();          // all waves done reading buf[cur] before reuse
  }

  // --- epilogue: bias + activation (+ residual), C layout M = r + 8g ---
#pragma unroll
  for (int mt = 0; mt < 2; ++mt) {
#pragma unroll
    for (int nt = 0; nt < 2; ++nt) {
      const int col = n0 + wn * 32 + nt * 16 + l16;
      const float bv = bias[col];
#pragma unroll
      for (int r = 0; r < 8; ++r) {
        const int row = m0 + wm * 32 + mt * 16 + r + 8 * g;
        float v = acc[mt][nt][r] + bv;
        if (ACT == 1) v = v / (1.f + __expf(-v));                       // SiLU
        else if (ACT == 2) v = 0.5f * v * (1.f + erff(v * 0.70710678118654752f)); // GELU
        const size_t idx = (size_t)row * N + col;
        if (RESID) v += resid[idx];
        if (OUT_BF16) outB[idx] = (bf16_t)v;
        else          outF[idx] = v;
      }
    }
  }
}

// ---------------------------------------------------------------------------
// Flash-style attention per (graph, head).  seq=256, DH=32, NH=8.
// qkv: (65536, 768) bf16 [q|k|v].  outB: (65536, 256) bf16 (o-proj input).
// K staged via async b128 copies (overlapped with manual V transpose);
// online softmax with ds_swizzle XOR reductions; P re-fragmented through
// per-wave LDS scratch.
// ---------------------------------------------------------------------------
#define KSTR 40    // Ks stride (bf16)
#define VSTR 264   // Vs stride (bf16): 528B rows, conflict-free
#define PSTR 40    // Ps stride (bf16)

__global__ __launch_bounds__(256) void attn_kernel(
    const bf16_t* __restrict__ qkv, bf16_t* __restrict__ outB) {
  __shared__ bf16_t Ks[256 * KSTR];       // [key][d]   (B frag for Q@K^T)
  __shared__ bf16_t Vs[32 * VSTR];        // [d][key]   (B frag for P@V)
  __shared__ bf16_t Ps[8 * 32 * PSTR];    // per-wave P scratch [row][key]

  const int bh = blockIdx.x;
  const int b = bh >> 3, h = bh & 7;
  const size_t atom0 = (size_t)b * 256;
  const int t = threadIdx.x;
  const int wave = t >> 5, lane = t & 31, g = lane >> 4, l16 = lane & 15;
  const float scale = 0.17677669529663687f;   // 1/sqrt(32)

  // async-stage K rows (64B per key row)
#pragma unroll
  for (int i = 0; i < 4; ++i) {
    const int p = i * 256 + t;
    const int n = p >> 2, seg = p & 3;
    async_copy_b128(qkv + (atom0 + n) * 768 + 256 + h * 32 + seg * 8,
                    &Ks[n * KSTR + seg * 8]);
  }
  // transposed stage of V (overlaps with async K copies)
#pragma unroll 4
  for (int it = 0; it < 32; ++it) {
    const int lin = it * 256 + t;
    const int n = lin >> 5, d = lin & 31;
    Vs[d * VSTR + n] = qkv[(atom0 + n) * 768 + 512 + h * 32 + d];
  }
  wait_async0();
  __syncthreads();

  // Q A-fragments: packed bf16 dwords straight from global
  FragB16 aq[2];
  const int r0 = wave * 32;
#pragma unroll
  for (int mt = 0; mt < 2; ++mt) {
    const size_t row = atom0 + r0 + mt * 16 + l16;
    const bf16_t* qp = qkv + row * 768 + h * 32;
#pragma unroll
    for (int j = 0; j < 8; ++j) {
      const int kb = 2 * (j & 3) + 16 * (j >> 2) + 8 * g;
      aq[mt].u[j] = *(const unsigned*)(qp + kb);
    }
  }

  float mrun[16], lrun[16];
#pragma unroll
  for (int i = 0; i < 16; ++i) { mrun[i] = -INFINITY; lrun[i] = 0.f; }
  v8f o[2][2];
#pragma unroll
  for (int i = 0; i < 2; ++i)
#pragma unroll
    for (int j = 0; j < 2; ++j) o[i][j] = v8f{0, 0, 0, 0, 0, 0, 0, 0};

  bf16_t* pw = Ps + wave * 32 * PSTR;
  const v8f zc = v8f{0, 0, 0, 0, 0, 0, 0, 0};

  for (int kb2 = 0; kb2 < 8; ++kb2) {
    // S = Q @ K^T for this 32-key block
    FragB16 bk[2];
#pragma unroll
    for (int nt = 0; nt < 2; ++nt) {
      const int key = kb2 * 32 + nt * 16 + l16;
#pragma unroll
      for (int j = 0; j < 8; ++j) {
        const int kd = 2 * (j & 3) + 16 * (j >> 2) + 8 * g;
        bk[nt].u[j] = *(const unsigned*)&Ks[key * KSTR + kd];
      }
    }
    v8f s[2][2];
#pragma unroll
    for (int mt = 0; mt < 2; ++mt)
#pragma unroll
      for (int nt = 0; nt < 2; ++nt)
        s[mt][nt] = wmma_bf16(aq[mt], bk[nt], zc);
#pragma unroll
    for (int mt = 0; mt < 2; ++mt)
#pragma unroll
      for (int nt = 0; nt < 2; ++nt)
#pragma unroll
        for (int r = 0; r < 8; ++r) s[mt][nt][r] *= scale;

    // online softmax (row groups of 16 lanes, ds_swizzle reductions)
#pragma unroll
    for (int mt = 0; mt < 2; ++mt) {
#pragma unroll
      for (int r = 0; r < 8; ++r) {
        const float mx = red_max16(fmaxf(s[mt][0][r], s[mt][1][r]));
        const int idx = mt * 8 + r;
        const float newm = fmaxf(mrun[idx], mx);
        const float corr = __expf(mrun[idx] - newm);
        mrun[idx] = newm;
        const float p0 = __expf(s[mt][0][r] - newm);
        const float p1 = __expf(s[mt][1][r] - newm);
        s[mt][0][r] = p0; s[mt][1][r] = p1;
        const float ps = red_add16(p0 + p1);
        lrun[idx] = lrun[idx] * corr + ps;
        o[mt][0][r] *= corr;
        o[mt][1][r] *= corr;
      }
    }

    // P -> wave-private LDS (C layout -> row major), then re-fragment as A
#pragma unroll
    for (int mt = 0; mt < 2; ++mt)
#pragma unroll
      for (int nt = 0; nt < 2; ++nt)
#pragma unroll
        for (int r = 0; r < 8; ++r)
          pw[(mt * 16 + r + 8 * g) * PSTR + nt * 16 + l16] = (bf16_t)s[mt][nt][r];
    asm volatile("s_wait_dscnt 0x0" ::: "memory");   // intra-wave LDS RAW

    FragB16 ap[2], bv[2];
#pragma unroll
    for (int mt = 0; mt < 2; ++mt)
#pragma unroll
      for (int j = 0; j < 8; ++j) {
        const int kb = 2 * (j & 3) + 16 * (j >> 2) + 8 * g;
        ap[mt].u[j] = *(const unsigned*)&pw[(mt * 16 + l16) * PSTR + kb];
      }
#pragma unroll
    for (int vj = 0; vj < 2; ++vj) {
      const int dcol = vj * 16 + l16;
#pragma unroll
      for (int j = 0; j < 8; ++j) {
        const int keyk = kb2 * 32 + 2 * (j & 3) + 16 * (j >> 2) + 8 * g;
        bv[vj].u[j] = *(const unsigned*)&Vs[dcol * VSTR + keyk];
      }
    }
#pragma unroll
    for (int mt = 0; mt < 2; ++mt)
#pragma unroll
      for (int vj = 0; vj < 2; ++vj)
        o[mt][vj] = wmma_bf16(ap[mt], bv[vj], o[mt][vj]);
  }

  // finalize: divide by row sum, write bf16 (input to o-projection GEMM)
#pragma unroll
  for (int mt = 0; mt < 2; ++mt)
#pragma unroll
    for (int vj = 0; vj < 2; ++vj)
#pragma unroll
      for (int r = 0; r < 8; ++r) {
        const int idx = mt * 8 + r;
        const float val = o[mt][vj][r] / lrun[idx];
        const size_t row = atom0 + r0 + mt * 16 + r + 8 * g;
        outB[row * 256 + h * 32 + vj * 16 + l16] = (bf16_t)val;
      }
}

// ---------------------------------------------------------------------------
// rowdot: out[row] = dot(A_bf16[row,0:256], w_f32) + b[0]. One row per wave.
// ---------------------------------------------------------------------------
__global__ __launch_bounds__(256) void rowdot_kernel(
    const bf16_t* __restrict__ A, const float* __restrict__ w,
    const float* __restrict__ bsc, float* __restrict__ out, int rows) {
  const int wave = threadIdx.x >> 5, lane = threadIdx.x & 31;
  const int row = blockIdx.x * 8 + wave;
  if (row >= rows) return;
  const bf16_t* ar = A + (size_t)row * 256;
  float s = 0.f;
#pragma unroll
  for (int i = 0; i < 8; ++i) s += (float)ar[lane + 32 * i] * w[lane + 32 * i];
  s = red_add32(s);
  if (lane == 0) out[row] = s + bsc[0];
}

// ---------------------------------------------------------------------------
// Softmax-attention pooling: one block per graph (256 atoms, 256 dims).
// ---------------------------------------------------------------------------
__global__ __launch_bounds__(256) void pool_kernel(
    const float* __restrict__ scores, const float* __restrict__ X,
    float* __restrict__ gemb) {
  __shared__ float red[256];
  __shared__ float w[256];
  const int b = blockIdx.x, t = threadIdx.x;
  const float s = scores[b * 256 + t];
  red[t] = s; __syncthreads();
  for (int off = 128; off > 0; off >>= 1) {
    if (t < off) red[t] = fmaxf(red[t], red[t + off]);
    __syncthreads();
  }
  const float mx = red[0]; __syncthreads();
  const float e = __expf(s - mx);
  red[t] = e; __syncthreads();
  for (int off = 128; off > 0; off >>= 1) {
    if (t < off) red[t] += red[t + off];
    __syncthreads();
  }
  const float inv = 1.f / red[0]; __syncthreads();
  w[t] = e * inv; __syncthreads();
  const float* xb = X + (size_t)b * 256 * 256;
  float acc = 0.f;
  for (int n = 0; n < 256; ++n) acc += w[n] * xb[n * 256 + t];
  gemb[b * 256 + t] = acc;
}

// ---------------------------------------------------------------------------
// Host-side orchestration
// ---------------------------------------------------------------------------
extern "C" void kernel_launch(void* const* d_in, const int* in_sizes, int n_in,
                              void* d_out, int out_size, void* d_ws, size_t ws_size,
                              hipStream_t stream) {
  (void)in_sizes; (void)n_in; (void)out_size; (void)ws_size;
  const int NA = 256 * 256;   // total atom rows

  const float* node_tokens = (const float*)d_in[0];
  const float* ln0_g = (const float*)d_in[2];
  const float* ln0_b = (const float*)d_in[3];
  const float* w_in1 = (const float*)d_in[4];
  const float* b_in1 = (const float*)d_in[5];
  const float* w_in2 = (const float*)d_in[6];
  const float* b_in2 = (const float*)d_in[7];
  const float* ln1_g = (const float*)d_in[8];
  const float* ln1_b = (const float*)d_in[9];
  const float* w_qkv = (const float*)d_in[10];
  const float* b_qkv = (const float*)d_in[11];
  const float* w_o   = (const float*)d_in[12];
  const float* b_o   = (const float*)d_in[13];
  const float* ln2_g = (const float*)d_in[14];
  const float* ln2_b = (const float*)d_in[15];
  const float* w_ff1 = (const float*)d_in[16];
  const float* b_ff1 = (const float*)d_in[17];
  const float* w_ff2 = (const float*)d_in[18];
  const float* b_ff2 = (const float*)d_in[19];
  const float* w_p1  = (const float*)d_in[20];
  const float* b_p1  = (const float*)d_in[21];
  const float* w_p2  = (const float*)d_in[22];
  const float* b_p2  = (const float*)d_in[23];
  const float* lnd_g = (const float*)d_in[24];
  const float* lnd_b = (const float*)d_in[25];
  const float* w_d1  = (const float*)d_in[26];
  const float* b_d1  = (const float*)d_in[27];
  const float* w_d2  = (const float*)d_in[28];
  const float* b_d2  = (const float*)d_in[29];
  const float* lne_g = (const float*)d_in[30];
  const float* lne_b = (const float*)d_in[31];
  const float* w_e1  = (const float*)d_in[32];
  const float* b_e1  = (const float*)d_in[33];
  const float* w_e2  = (const float*)d_in[34];
  const float* b_e2  = (const float*)d_in[35];

  char* wp = (char*)d_ws;
  auto alloc = [&](size_t bytes) -> void* {
    void* p = (void*)wp; wp += (bytes + 255) & ~(size_t)255; return p;
  };
  // transposed bf16 weight copies: (L,K,N) f32 -> (L,N,K) bf16
  auto cvtT = [&](const float* src, int K, int N, int L) -> bf16_t* {
    long long total = (long long)L * K * N;
    bf16_t* dst = (bf16_t*)alloc((size_t)total * sizeof(bf16_t));
    int grid = (int)((total + 2047) / 2048); if (grid > 4096) grid = 4096; if (grid < 1) grid = 1;
    cvtT_f32_bf16<<<grid, 256, 0, stream>>>(src, dst, K, N, total);
    return dst;
  };

  bf16_t* cw_in1 = cvtT(w_in1, 256, 256, 1);
  bf16_t* cw_in2 = cvtT(w_in2, 256, 256, 1);
  bf16_t* cw_qkv = cvtT(w_qkv, 256, 768, 2);
  bf16_t* cw_o   = cvtT(w_o,   256, 256, 2);
  bf16_t* cw_ff1 = cvtT(w_ff1, 256, 1024, 2);
  bf16_t* cw_ff2 = cvtT(w_ff2, 1024, 256, 2);
  bf16_t* cw_p1  = cvtT(w_p1,  256, 256, 1);
  bf16_t* cw_d1  = cvtT(w_d1,  256, 128, 1);
  bf16_t* cw_d2  = cvtT(w_d2,  128, 128, 1);
  bf16_t* cw_e1  = cvtT(w_e1,  128, 256, 1);

  // activation buffers
  bf16_t* bufH  = (bf16_t*)alloc((size_t)NA * 256  * sizeof(bf16_t));
  bf16_t* bufF  = (bf16_t*)alloc((size_t)NA * 1024 * sizeof(bf16_t));
  bf16_t* qkvB  = (bf16_t*)alloc((size_t)NA * 768  * sizeof(bf16_t));
  float*  X     = (float*) alloc((size_t)NA * 256  * sizeof(float));
  float*  scores = (float*)alloc((size_t)NA * sizeof(float));
  float*  gemb   = (float*)alloc(256 * 256 * sizeof(float));
  bf16_t* gbf    = (bf16_t*)alloc(256 * 256 * sizeof(bf16_t));
  bf16_t* d1buf  = (bf16_t*)alloc(256 * 128 * sizeof(bf16_t));
  bf16_t* ebf    = (bf16_t*)alloc(256 * 128 * sizeof(bf16_t));
  bf16_t* e1buf  = (bf16_t*)alloc(256 * 256 * sizeof(bf16_t));

  float* desc_out   = (float*)d_out;              // (256,128)
  float* energy_out = (float*)d_out + 256 * 128;  // (256,)

  auto gemmN = [&](const bf16_t* A, const bf16_t* W, const float* bias,
                   float* outF, int M, int Nn, int K) {
    gemm_bf16_kernel<0, false, false><<<dim3(Nn / BN, M / BM), 256, 0, stream>>>(
        A, W, bias, nullptr, outF, nullptr, M, Nn, K);
  };
  auto gemmB = [&](const bf16_t* A, const bf16_t* W, const float* bias,
                   bf16_t* outB, int M, int Nn, int K) {
    gemm_bf16_kernel<0, true, false><<<dim3(Nn / BN, M / BM), 256, 0, stream>>>(
        A, W, bias, nullptr, nullptr, outB, M, Nn, K);
  };
  auto gemmR = [&](const bf16_t* A, const bf16_t* W, const float* bias,
                   const float* resid, float* outF, int M, int Nn, int K) {
    gemm_bf16_kernel<0, false, true><<<dim3(Nn / BN, M / BM), 256, 0, stream>>>(
        A, W, bias, resid, outF, nullptr, M, Nn, K);
  };
  auto gemmS = [&](const bf16_t* A, const bf16_t* W, const float* bias,
                   bf16_t* outB, int M, int Nn, int K) {
    gemm_bf16_kernel<1, true, false><<<dim3(Nn / BN, M / BM), 256, 0, stream>>>(
        A, W, bias, nullptr, nullptr, outB, M, Nn, K);
  };
  auto gemmG = [&](const bf16_t* A, const bf16_t* W, const float* bias,
                   bf16_t* outB, int M, int Nn, int K) {
    gemm_bf16_kernel<2, true, false><<<dim3(Nn / BN, M / BM), 256, 0, stream>>>(
        A, W, bias, nullptr, nullptr, outB, M, Nn, K);
  };

  // ---- input projection ----
  ln_rows_kernel<<<NA / 8, 256, 0, stream>>>(node_tokens, ln0_g, ln0_b, bufH, 256, NA);
  gemmS(bufH, cw_in1, b_in1, bufF, NA, 256, 256);          // SiLU(x@w_in1+b)
  gemmN(bufF, cw_in2, b_in2, X,   NA, 256, 256);           // -> X (f32 residual)

  // ---- transformer layers ----
  for (int d = 0; d < 2; ++d) {
    ln_rows_kernel<<<NA / 8, 256, 0, stream>>>(X, ln1_g + d * 256, ln1_b + d * 256,
                                               bufH, 256, NA);
    gemmB(bufH, cw_qkv + (size_t)d * 256 * 768, b_qkv + d * 768, qkvB, NA, 768, 256);
    attn_kernel<<<256 * 8, 256, 0, stream>>>(qkvB, bufH);
    gemmR(bufH, cw_o + (size_t)d * 256 * 256, b_o + d * 256, X, X, NA, 256, 256);

    ln_rows_kernel<<<NA / 8, 256, 0, stream>>>(X, ln2_g + d * 256, ln2_b + d * 256,
                                               bufH, 256, NA);
    gemmG(bufH, cw_ff1 + (size_t)d * 256 * 1024, b_ff1 + d * 1024, bufF, NA, 1024, 256);
    gemmR(bufF, cw_ff2 + (size_t)d * 1024 * 256, b_ff2 + d * 256, X, X, NA, 256, 1024);
  }

  // ---- pooling ----
  cvt_f32_bf16<<<4096, 256, 0, stream>>>(X, bufH, (long long)NA * 256);
  gemmS(bufH, cw_p1, b_p1, bufF, NA, 256, 256);
  rowdot_kernel<<<NA / 8, 256, 0, stream>>>(bufF, w_p2, b_p2, scores, NA);
  pool_kernel<<<256, 256, 0, stream>>>(scores, X, gemb);

  // ---- descriptor head ----
  ln_rows_kernel<<<256 / 8, 256, 0, stream>>>(gemb, lnd_g, lnd_b, gbf, 256, 256);
  gemmS(gbf, cw_d1, b_d1, d1buf, 256, 128, 256);
  gemmN(d1buf, cw_d2, b_d2, desc_out, 256, 128, 128);

  // ---- energy head ----
  ln_rows_kernel<<<256 / 8, 256, 0, stream>>>(desc_out, lne_g, lne_b, ebf, 128, 256);
  gemmS(ebf, cw_e1, b_e1, e1buf, 256, 256, 128);
  rowdot_kernel<<<256 / 8, 256, 0, stream>>>(e1buf, w_e2, b_e2, energy_out, 256);
}
